// TripleAttention_63161789055236
// MI455X (gfx1250) — matrix operand order repaired and checked by reading, not compiled, not run on hardware
//
#include <hip/hip_runtime.h>

#define NEGV (-1000000000.0f)
#define BATCH 8
#define LP 1024
#define JQ 128
#define NA 256
#define DD 512
#define GC 4608

typedef __attribute__((ext_vector_type(16))) __bf16 v16bf;
typedef __attribute__((ext_vector_type(8)))  float  v8f;
typedef __attribute__((ext_vector_type(4)))  unsigned int v4u;

union FragU { v4u q[2]; v16bf v; };
union Pack8 { v4u q; unsigned short s[8]; };

__device__ __forceinline__ unsigned short f2bf(float f) {
  union { float f; unsigned int u; } x; x.f = f;
  unsigned int r = x.u + 0x7FFFu + ((x.u >> 16) & 1u);
  return (unsigned short)(r >> 16);
}

// 2x2 register-blocked WMMA macro-tile: 32x32 of C = A(MxK)*B(NxK)^T, bf16 in, f32 accum.
// Fragment striping per CDNA5 ISA (16-bit operands, wave32):
//   A: lane (m=lane&15): K-chunks [k + (lane<16?0:8), +8) and [k+16+(lane<16?0:8), +8)
//   B: lane (n=lane&15): K-chunks [k + (lane<16?0:8), +8) and [k+16+(lane<16?0:8), +8)
__device__ __forceinline__ void gemm_tile_nt_2x2(const unsigned short* __restrict__ A,
                                                 const unsigned short* __restrict__ Bm,
                                                 int K, int row0, int col0, int lane,
                                                 v8f acc[4]) {
  const int half = lane >> 4;
  const int lr   = lane & 15;
  const unsigned short* ap0 = A  + (size_t)(row0 + lr) * K;
  const unsigned short* ap1 = A  + (size_t)(row0 + 16 + lr) * K;
  const unsigned short* bp0 = Bm + (size_t)(col0 + lr) * K;
  const unsigned short* bp1 = Bm + (size_t)(col0 + 16 + lr) * K;
  const int aoff = half << 3;   // 0 or 8 elements
  const int boff = half << 3;   // 0 or 8 elements
  for (int k = 0; k < K; k += 32) {
    FragU a0, a1, b0, b1;
    const v4u* pa0 = (const v4u*)(ap0 + k + aoff);
    const v4u* pa1 = (const v4u*)(ap1 + k + aoff);
    const v4u* pb0 = (const v4u*)(bp0 + k + boff);
    const v4u* pb1 = (const v4u*)(bp1 + k + boff);
    a0.q[0] = pa0[0]; a0.q[1] = pa0[2];     // +16 bf16 elements
    a1.q[0] = pa1[0]; a1.q[1] = pa1[2];
    b0.q[0] = pb0[0]; b0.q[1] = pb0[2];
    b1.q[0] = pb1[0]; b1.q[1] = pb1[2];
    // 4 independent accumulators -> WMMAs issue back-to-back (no D->A/B hazard)
    acc[0] = __builtin_amdgcn_wmma_f32_16x16x32_bf16(false, a0.v, false, b0.v, (short)0, acc[0], false, false);
    acc[1] = __builtin_amdgcn_wmma_f32_16x16x32_bf16(false, a0.v, false, b1.v, (short)0, acc[1], false, false);
    acc[2] = __builtin_amdgcn_wmma_f32_16x16x32_bf16(false, a1.v, false, b0.v, (short)0, acc[2], false, false);
    acc[3] = __builtin_amdgcn_wmma_f32_16x16x32_bf16(false, a1.v, false, b1.v, (short)0, acc[3], false, false);
  }
}

// ---------------- stage kernels ----------------

__global__ void scale_cvt_kernel(const float* __restrict__ src, const float* __restrict__ w3,
                                 unsigned short* __restrict__ dst, size_t total) {
  size_t i = (size_t)blockIdx.x * blockDim.x + threadIdx.x;
  if (i >= total) return;
  dst[i] = f2bf(src[i] * w3[i & (DD - 1)]);
}

__global__ void cvt_kernel(const float* __restrict__ src,
                           unsigned short* __restrict__ dst, size_t total) {
  size_t i = (size_t)blockIdx.x * blockDim.x + threadIdx.x;
  if (i >= total) return;
  dst[i] = f2bf(src[i]);
}

// (B,R,D) f32 -> (B,D,R) bf16, 32x32 LDS tiles
__global__ void transpose_cvt_kernel(const float* __restrict__ src,
                                     unsigned short* __restrict__ dst, int R) {
  __shared__ float t[32][33];
  const int b  = blockIdx.z;
  const int r0 = blockIdx.x << 5;
  const int c0 = blockIdx.y << 5;
  #pragma unroll
  for (int i = 0; i < 4; ++i) {
    int r = threadIdx.y + (i << 3);
    t[r][threadIdx.x] = src[((size_t)b * R + r0 + r) * DD + c0 + threadIdx.x];
  }
  __syncthreads();
  #pragma unroll
  for (int i = 0; i < 4; ++i) {
    int c = threadIdx.y + (i << 3);
    dst[((size_t)b * DD + c0 + c) * R + r0 + threadIdx.x] = f2bf(t[threadIdx.x][c]);
  }
}

__global__ void rowdot_kernel(const float* __restrict__ E, const float* __restrict__ wv,
                              float* __restrict__ out, int rows) {
  int w = blockIdx.x * (blockDim.x >> 5) + (threadIdx.x >> 5);
  if (w >= rows) return;
  int lane = threadIdx.x & 31;
  const float* e = E + (size_t)w * DD;
  float s = 0.f;
  for (int j = lane; j < DD; j += 32) s += e[j] * wv[j];
  for (int off = 16; off; off >>= 1) s += __shfl_xor(s, off, 32);
  if (lane == 0) out[w] = s;
}

// U = A*B^T + rx + cy, masked
__global__ void gemm_u_kernel(const unsigned short* __restrict__ A,
                              const unsigned short* __restrict__ Bm,
                              const float* __restrict__ rx, const float* __restrict__ cy,
                              const unsigned char* __restrict__ mx,
                              const unsigned char* __restrict__ my,
                              float* __restrict__ U, int M, int N, int K) {
  const int lane    = threadIdx.x & 31;
  const int tilesN  = N >> 5;
  const int tilesPB = (M >> 5) * tilesN;
  const int w = blockIdx.x * (blockDim.x >> 5) + (threadIdx.x >> 5);
  if (w >= BATCH * tilesPB) return;
  const int b = w / tilesPB;
  const int t = w - b * tilesPB;
  const int row0 = (t / tilesN) << 5;
  const int col0 = (t % tilesN) << 5;
  v8f acc[4] = {v8f{}, v8f{}, v8f{}, v8f{}};
  gemm_tile_nt_2x2(A + (size_t)b * M * K, Bm + (size_t)b * N * K, K, row0, col0, lane, acc);
  const int half = lane >> 4, lr = lane & 15;
  float* urow = U + (size_t)b * M * N;
  #pragma unroll
  for (int i = 0; i < 2; ++i) {
    #pragma unroll
    for (int j = 0; j < 2; ++j) {
      const v8f a = acc[i * 2 + j];
      const int n = col0 + (j << 4) + lr;
      const float cyv = cy[b * N + n];
      const bool okc = my[b * N + n] != 0;
      #pragma unroll
      for (int r = 0; r < 8; ++r) {
        int m = row0 + (i << 4) + (half << 3) + r;
        float v = a[r] + rx[b * M + m] + cyv;
        urow[(size_t)m * N + n] = (okc && mx[b * M + m]) ? v : NEGV;
      }
    }
  }
}

__global__ void softmax_row_kernel(const float* __restrict__ U,
                                   unsigned short* __restrict__ Ab, int rows, int NC) {
  int w = blockIdx.x * (blockDim.x >> 5) + (threadIdx.x >> 5);
  if (w >= rows) return;
  int lane = threadIdx.x & 31;
  const float* u = U + (size_t)w * NC;
  const int cnt = NC >> 5;  // 4 or 8
  float e[8];
  float mx = -3.4e38f;
  for (int i = 0; i < cnt; ++i) mx = fmaxf(mx, u[lane + (i << 5)]);
  for (int off = 16; off; off >>= 1) mx = fmaxf(mx, __shfl_xor(mx, off, 32));
  float s = 0.f;
  for (int i = 0; i < cnt; ++i) { e[i] = __expf(u[lane + (i << 5)] - mx); s += e[i]; }
  for (int off = 16; off; off >>= 1) s += __shfl_xor(s, off, 32);
  float inv = 1.0f / s;
  unsigned short* o = Ab + (size_t)w * NC;
  for (int i = 0; i < cnt; ++i) o[lane + (i << 5)] = f2bf(e[i] * inv);
}

// col-softmax of U (B,R,NC) over R, written transposed: Bt (B,NC,R) bf16
__global__ void softmax_col_kernel(const float* __restrict__ U,
                                   unsigned short* __restrict__ Bt, int R, int NC) {
  int w = blockIdx.x * (blockDim.x >> 5) + (threadIdx.x >> 5);
  if (w >= BATCH * NC) return;
  int lane = threadIdx.x & 31;
  int b = w / NC, col = w - b * NC;
  const float* u = U + (size_t)b * R * NC + col;
  float mx = -3.4e38f;
  for (int r = lane; r < R; r += 32) mx = fmaxf(mx, u[(size_t)r * NC]);
  for (int off = 16; off; off >>= 1) mx = fmaxf(mx, __shfl_xor(mx, off, 32));
  float s = 0.f;
  for (int r = lane; r < R; r += 32) s += __expf(u[(size_t)r * NC] - mx);
  for (int off = 16; off; off >>= 1) s += __shfl_xor(s, off, 32);
  float inv = 1.0f / s;
  unsigned short* o = Bt + ((size_t)b * NC + col) * R;
  for (int r = lane; r < R; r += 32) o[r] = f2bf(__expf(u[(size_t)r * NC] - mx) * inv);
}

// bar = A(MxK)*BT(512xK)^T ; writes f32 (Mx512) and bf16 transposed (512xM)
__global__ void gemm_bar_kernel(const unsigned short* __restrict__ A,
                                const unsigned short* __restrict__ BT,
                                float* __restrict__ C, unsigned short* __restrict__ CT,
                                int M, int K) {
  const int lane    = threadIdx.x & 31;
  const int tilesN  = DD >> 5;  // 16
  const int tilesPB = (M >> 5) * tilesN;
  const int w = blockIdx.x * (blockDim.x >> 5) + (threadIdx.x >> 5);
  if (w >= BATCH * tilesPB) return;
  const int b = w / tilesPB;
  const int t = w - b * tilesPB;
  const int row0 = (t / tilesN) << 5;
  const int col0 = (t % tilesN) << 5;
  v8f acc[4] = {v8f{}, v8f{}, v8f{}, v8f{}};
  gemm_tile_nt_2x2(A + (size_t)b * M * K, BT + (size_t)b * DD * K, K, row0, col0, lane, acc);
  const int half = lane >> 4, lr = lane & 15;
  float* crow = C + (size_t)b * M * DD;
  #pragma unroll
  for (int i = 0; i < 2; ++i) {
    #pragma unroll
    for (int j = 0; j < 2; ++j) {
      const v8f a = acc[i * 2 + j];
      const int n = col0 + (j << 4) + lr;
      const int mb = row0 + (i << 4) + (half << 3);
      Pack8 p;
      #pragma unroll
      for (int r = 0; r < 8; ++r) {
        crow[(size_t)(mb + r) * DD + n] = a[r];
        p.s[r] = f2bf(a[r]);
      }
      *(v4u*)(CT + ((size_t)b * DD + n) * M + mb) = p.q;
    }
  }
}

// Cslice = A(MxK)*BT(512xK)^T -> G[b*M+m][sliceOff+n], row stride 4608
__global__ void gemm_out_kernel(const unsigned short* __restrict__ A,
                                const unsigned short* __restrict__ BT,
                                float* __restrict__ G, int sliceOff, int M, int K) {
  const int lane    = threadIdx.x & 31;
  const int tilesN  = DD >> 5;
  const int tilesPB = (M >> 5) * tilesN;
  const int w = blockIdx.x * (blockDim.x >> 5) + (threadIdx.x >> 5);
  if (w >= BATCH * tilesPB) return;
  const int b = w / tilesPB;
  const int t = w - b * tilesPB;
  const int row0 = (t / tilesN) << 5;
  const int col0 = (t % tilesN) << 5;
  v8f acc[4] = {v8f{}, v8f{}, v8f{}, v8f{}};
  gemm_tile_nt_2x2(A + (size_t)b * M * K, BT + (size_t)b * DD * K, K, row0, col0, lane, acc);
  const int half = lane >> 4, lr = lane & 15;
  float* g = G + (size_t)b * M * GC + sliceOff;
  #pragma unroll
  for (int i = 0; i < 2; ++i) {
    #pragma unroll
    for (int j = 0; j < 2; ++j) {
      const v8f a = acc[i * 2 + j];
      const int n = col0 + (j << 4) + lr;
      const int mb = row0 + (i << 4) + (half << 3);
      #pragma unroll
      for (int r = 0; r < 8; ++r) g[(size_t)(mb + r) * GC + n] = a[r];
    }
  }
}

// slices 0..4 of a G block: [E, bar1, bar2, E*bar1, E*bar2]
__global__ void ew_out_kernel(const float* __restrict__ E,
                              const float* __restrict__ bar1, const float* __restrict__ bar2,
                              float* __restrict__ G, size_t rowsTotal) {
  size_t i = (size_t)blockIdx.x * blockDim.x + threadIdx.x;
  if (i >= rowsTotal * DD) return;
  size_t row = i >> 9;
  int    col = (int)(i & (DD - 1));
  float e = E[i], b1 = bar1[i], b2 = bar2[i];
  float* g = G + row * GC + col;
  g[0]        = e;
  g[DD]       = b1;
  g[2 * DD]   = b2;
  g[3 * DD]   = e * b1;
  g[4 * DD]   = e * b2;
}

// ---------------- host launch ----------------

extern "C" void kernel_launch(void* const* d_in, const int* in_sizes, int n_in,
                              void* d_out, int out_size, void* d_ws, size_t ws_size,
                              hipStream_t stream) {
  (void)in_sizes; (void)n_in; (void)out_size; (void)ws_size;
  const float* Ep = (const float*)d_in[0];
  const float* Eq = (const float*)d_in[1];
  const float* Ea = (const float*)d_in[2];
  const unsigned char* mp = (const unsigned char*)d_in[3];
  const unsigned char* mq = (const unsigned char*)d_in[4];
  const unsigned char* ma = (const unsigned char*)d_in[5];
  const float* w_pq = (const float*)d_in[6];
  const float* w_pa = (const float*)d_in[7];
  const float* w_qa = (const float*)d_in[8];

  float* Gp = (float*)d_out;
  float* Gq = Gp + (size_t)BATCH * LP * GC;
  float* Ga = Gq + (size_t)BATCH * JQ * GC;

  char* wsb = (char*)d_ws;
  size_t off = 0;
  auto alloc = [&](size_t bytes) { size_t o = off; off = (off + bytes + 255) & ~(size_t)255; return (void*)(wsb + o); };
  auto bf = [&](size_t n) { return (unsigned short*)alloc(n * 2); };
  auto fp = [&](size_t n) { return (float*)alloc(n * 4); };

  unsigned short* XpqB = bf((size_t)BATCH * LP * DD);
  unsigned short* XpaB = bf((size_t)BATCH * LP * DD);
  unsigned short* XqaB = bf((size_t)BATCH * JQ * DD);
  unsigned short* EqB  = bf((size_t)BATCH * JQ * DD);
  unsigned short* EaB  = bf((size_t)BATCH * NA * DD);
  unsigned short* EpT  = bf((size_t)BATCH * DD * LP);
  unsigned short* EqT  = bf((size_t)BATCH * DD * JQ);
  unsigned short* EaT  = bf((size_t)BATCH * DD * NA);
  float* rp_pq = fp((size_t)BATCH * LP);
  float* rp_pa = fp((size_t)BATCH * LP);
  float* cq_pq = fp((size_t)BATCH * JQ);
  float* rq_qa = fp((size_t)BATCH * JQ);
  float* ca_pa = fp((size_t)BATCH * NA);
  float* ca_qa = fp((size_t)BATCH * NA);
  float* Upq = fp((size_t)BATCH * LP * JQ);
  float* Upa = fp((size_t)BATCH * LP * NA);
  float* Uqa = fp((size_t)BATCH * JQ * NA);
  unsigned short* ApqB = bf((size_t)BATCH * LP * JQ);
  unsigned short* ApaB = bf((size_t)BATCH * LP * NA);
  unsigned short* AqaB = bf((size_t)BATCH * JQ * NA);
  unsigned short* BpqB = bf((size_t)BATCH * JQ * LP);
  unsigned short* BpaB = bf((size_t)BATCH * NA * LP);
  unsigned short* BqaB = bf((size_t)BATCH * NA * JQ);
  float* Apqb = fp((size_t)BATCH * LP * DD);
  float* Apab = fp((size_t)BATCH * LP * DD);
  float* Bpqb = fp((size_t)BATCH * JQ * DD);
  float* Bpab = fp((size_t)BATCH * NA * DD);
  float* Aqab = fp((size_t)BATCH * JQ * DD);
  float* Bqab = fp((size_t)BATCH * NA * DD);
  unsigned short* ApqbT = bf((size_t)BATCH * DD * LP);
  unsigned short* ApabT = bf((size_t)BATCH * DD * LP);
  unsigned short* BpqbT = bf((size_t)BATCH * DD * JQ);
  unsigned short* BpabT = bf((size_t)BATCH * DD * NA);
  unsigned short* AqabT = bf((size_t)BATCH * DD * JQ);
  unsigned short* BqabT = bf((size_t)BATCH * DD * NA);

  auto blocksFor = [](size_t n, int per) { return (unsigned)((n + per - 1) / per); };

  // stage 1: converted / scaled operands
  size_t tLP = (size_t)BATCH * LP * DD, tJQ = (size_t)BATCH * JQ * DD, tNA = (size_t)BATCH * NA * DD;
  scale_cvt_kernel<<<blocksFor(tLP, 256), 256, 0, stream>>>(Ep, w_pq + 2 * DD, XpqB, tLP);
  scale_cvt_kernel<<<blocksFor(tLP, 256), 256, 0, stream>>>(Ep, w_pa + 2 * DD, XpaB, tLP);
  scale_cvt_kernel<<<blocksFor(tJQ, 256), 256, 0, stream>>>(Eq, w_qa + 2 * DD, XqaB, tJQ);
  cvt_kernel<<<blocksFor(tJQ, 256), 256, 0, stream>>>(Eq, EqB, tJQ);
  cvt_kernel<<<blocksFor(tNA, 256), 256, 0, stream>>>(Ea, EaB, tNA);
  transpose_cvt_kernel<<<dim3(LP / 32, DD / 32, BATCH), dim3(32, 8), 0, stream>>>(Ep, EpT, LP);
  transpose_cvt_kernel<<<dim3(JQ / 32, DD / 32, BATCH), dim3(32, 8), 0, stream>>>(Eq, EqT, JQ);
  transpose_cvt_kernel<<<dim3(NA / 32, DD / 32, BATCH), dim3(32, 8), 0, stream>>>(Ea, EaT, NA);

  // stage 2: row/col bias vectors
  rowdot_kernel<<<blocksFor(BATCH * LP, 8), 256, 0, stream>>>(Ep, w_pq, rp_pq, BATCH * LP);
  rowdot_kernel<<<blocksFor(BATCH * JQ, 8), 256, 0, stream>>>(Eq, w_pq + DD, cq_pq, BATCH * JQ);
  rowdot_kernel<<<blocksFor(BATCH * LP, 8), 256, 0, stream>>>(Ep, w_pa, rp_pa, BATCH * LP);
  rowdot_kernel<<<blocksFor(BATCH * NA, 8), 256, 0, stream>>>(Ea, w_pa + DD, ca_pa, BATCH * NA);
  rowdot_kernel<<<blocksFor(BATCH * JQ, 8), 256, 0, stream>>>(Eq, w_qa, rq_qa, BATCH * JQ);
  rowdot_kernel<<<blocksFor(BATCH * NA, 8), 256, 0, stream>>>(Ea, w_qa + DD, ca_qa, BATCH * NA);

  // stage 3: masked similarity logits (WMMA), 32x32 macro-tiles
  auto gtiles = [](int M, int N) { return (size_t)BATCH * (M >> 5) * (N >> 5); };
  gemm_u_kernel<<<blocksFor(gtiles(LP, JQ), 8), 256, 0, stream>>>(XpqB, EqB, rp_pq, cq_pq, mp, mq, Upq, LP, JQ, DD);
  gemm_u_kernel<<<blocksFor(gtiles(LP, NA), 8), 256, 0, stream>>>(XpaB, EaB, rp_pa, ca_pa, mp, ma, Upa, LP, NA, DD);
  gemm_u_kernel<<<blocksFor(gtiles(JQ, NA), 8), 256, 0, stream>>>(XqaB, EaB, rq_qa, ca_qa, mq, ma, Uqa, JQ, NA, DD);

  // stage 4: softmaxes
  softmax_row_kernel<<<blocksFor(BATCH * LP, 8), 256, 0, stream>>>(Upq, ApqB, BATCH * LP, JQ);
  softmax_row_kernel<<<blocksFor(BATCH * LP, 8), 256, 0, stream>>>(Upa, ApaB, BATCH * LP, NA);
  softmax_row_kernel<<<blocksFor(BATCH * JQ, 8), 256, 0, stream>>>(Uqa, AqaB, BATCH * JQ, NA);
  softmax_col_kernel<<<blocksFor((size_t)BATCH * JQ, 8), 256, 0, stream>>>(Upq, BpqB, LP, JQ);
  softmax_col_kernel<<<blocksFor((size_t)BATCH * NA, 8), 256, 0, stream>>>(Upa, BpaB, LP, NA);
  softmax_col_kernel<<<blocksFor((size_t)BATCH * NA, 8), 256, 0, stream>>>(Uqa, BqaB, JQ, NA);

  // stage 5: bar GEMMs (WMMA), f32 + transposed bf16 epilogue
  gemm_bar_kernel<<<blocksFor(gtiles(LP, DD), 8), 256, 0, stream>>>(ApqB, EqT, Apqb, ApqbT, LP, JQ);
  gemm_bar_kernel<<<blocksFor(gtiles(JQ, DD), 8), 256, 0, stream>>>(BpqB, EpT, Bpqb, BpqbT, JQ, LP);
  gemm_bar_kernel<<<blocksFor(gtiles(LP, DD), 8), 256, 0, stream>>>(ApaB, EaT, Apab, ApabT, LP, NA);
  gemm_bar_kernel<<<blocksFor(gtiles(NA, DD), 8), 256, 0, stream>>>(BpaB, EpT, Bpab, BpabT, NA, LP);
  gemm_bar_kernel<<<blocksFor(gtiles(JQ, DD), 8), 256, 0, stream>>>(AqaB, EaT, Aqab, AqabT, JQ, NA);
  gemm_bar_kernel<<<blocksFor(gtiles(NA, DD), 8), 256, 0, stream>>>(BqaB, EqT, Bqab, BqabT, NA, JQ);

  // stage 6: elementwise slices 0..4
  ew_out_kernel<<<blocksFor(tLP, 256), 256, 0, stream>>>(Ep, Apqb, Apab, Gp, (size_t)BATCH * LP);
  ew_out_kernel<<<blocksFor(tJQ, 256), 256, 0, stream>>>(Eq, Bpqb, Aqab, Gq, (size_t)BATCH * JQ);
  ew_out_kernel<<<blocksFor(tNA, 256), 256, 0, stream>>>(Ea, Bpab, Bqab, Ga, (size_t)BATCH * NA);

  // stage 7: second-level GEMMs (WMMA) into G slices 5..8
  gemm_out_kernel<<<blocksFor(gtiles(LP, DD), 8), 256, 0, stream>>>(ApqB, BpqbT, Gp, 5 * DD, LP, JQ);
  gemm_out_kernel<<<blocksFor(gtiles(LP, DD), 8), 256, 0, stream>>>(ApqB, AqabT, Gp, 6 * DD, LP, JQ);
  gemm_out_kernel<<<blocksFor(gtiles(LP, DD), 8), 256, 0, stream>>>(ApaB, BpabT, Gp, 7 * DD, LP, NA);
  gemm_out_kernel<<<blocksFor(gtiles(LP, DD), 8), 256, 0, stream>>>(ApaB, BqabT, Gp, 8 * DD, LP, NA);
  gemm_out_kernel<<<blocksFor(gtiles(JQ, DD), 8), 256, 0, stream>>>(BpqB, ApqbT, Gq, 5 * DD, JQ, LP);
  gemm_out_kernel<<<blocksFor(gtiles(JQ, DD), 8), 256, 0, stream>>>(BpqB, ApabT, Gq, 6 * DD, JQ, LP);
  gemm_out_kernel<<<blocksFor(gtiles(JQ, DD), 8), 256, 0, stream>>>(AqaB, BpabT, Gq, 7 * DD, JQ, NA);
  gemm_out_kernel<<<blocksFor(gtiles(JQ, DD), 8), 256, 0, stream>>>(AqaB, BqabT, Gq, 8 * DD, JQ, NA);
  gemm_out_kernel<<<blocksFor(gtiles(NA, DD), 8), 256, 0, stream>>>(BpaB, ApqbT, Ga, 5 * DD, NA, LP);
  gemm_out_kernel<<<blocksFor(gtiles(NA, DD), 8), 256, 0, stream>>>(BpaB, ApabT, Ga, 6 * DD, NA, LP);
  gemm_out_kernel<<<blocksFor(gtiles(NA, DD), 8), 256, 0, stream>>>(BqaB, BpqbT, Ga, 7 * DD, NA, JQ);
  gemm_out_kernel<<<blocksFor(gtiles(NA, DD), 8), 256, 0, stream>>>(BqaB, AqabT, Ga, 8 * DD, NA, JQ);
}
